// AudioEncoder_66468913873514
// MI455X (gfx1250) — matrix operand order repaired and checked by reading, not compiled, not run on hardware
//
#include <hip/hip_runtime.h>

// ---------------------------------------------------------------------------
// Whisper-style audio encoder for MI455X (gfx1250, wave32, WMMA).
// All matmuls (convs via im2col / overlapping-row trick, QKV/out/MLP,
// QK^T and PV) run through v_wmma_f32_16x16x32_bf16 with f32 accumulation.
// Block tile 128x128, 8 waves in 4(M)x2(N) grid, wave tile 32x64 (2x4 frags):
// 8 WMMAs per 12 ds_load_b128 per K-step.  Workspace requirement: ~400 MB.
// ---------------------------------------------------------------------------

typedef __attribute__((ext_vector_type(16))) __bf16        bf16x16;
typedef __attribute__((ext_vector_type(8)))  float         f32x8;
typedef __attribute__((ext_vector_type(4)))  unsigned int  u32x4;

#define FLAG_BIAS   1
#define FLAG_GELU   2
#define FLAG_STF32  4
#define FLAG_ADDF32 8

__device__ __forceinline__ unsigned short f2bf(float f) {
  unsigned int u = __float_as_uint(f);
  u += 0x7FFFu + ((u >> 16) & 1u);            // round-to-nearest-even
  return (unsigned short)(u >> 16);
}
__device__ __forceinline__ float bf2f(unsigned short h) {
  return __uint_as_float(((unsigned int)h) << 16);
}
__device__ __forceinline__ float gelu_exact(float x) {
  return 0.5f * x * (1.0f + erff(x * 0.70710678118654752f));
}

union FragU { bf16x16 v; u32x4 q[2]; };

// ---------------------------------------------------------------------------
// Generic batched bf16 GEMM:  C[z](M x N) = A[z](M x K, lda) * B[z](N x K, ldb)^T
// Output offset: cOff = (z / cInnerN)*cOuter + (z % cInnerN)*cInner, then
// row*ldc + col.  flags: bias / gelu / store-f32 / add-f32 (else store bf16).
// ---------------------------------------------------------------------------
__global__ __launch_bounds__(256) void gemm_bf16_kernel(
    const unsigned short* __restrict__ A, long aStride, int lda,
    const unsigned short* __restrict__ Bw, long bStride, int ldb,
    float* Cf, unsigned short* Cb,
    long cOuter, long cInner, int cInnerN, int ldc,
    int K, const float* __restrict__ bias, int flags)
{
  __shared__ __align__(16) unsigned short sA[128 * 40];   // 128 rows x 32 halves (+8 pad)
  __shared__ __align__(16) unsigned short sB[128 * 40];   // 128 rows x 32 halves (+8 pad)

  const int tid  = threadIdx.x;
  const int lane = tid & 31;
  const int wave = tid >> 5;
  const int wm   = wave & 3;         // 4 wave-rows  x 32 M
  const int wn   = wave >> 2;        // 2 wave-cols  x 64 N
  const int hi   = lane >> 4;        // half-wave select (lanes 16..31)
  const int l15  = lane & 15;
  const int z    = blockIdx.z;

  const unsigned short* Ab = A  + (long)z * aStride + (long)blockIdx.y * 128 * lda;
  const unsigned short* Bb = Bw + (long)z * bStride + (long)blockIdx.x * 128 * ldb;

  const int trow = tid >> 1;         // 0..127, two threads per tile row
  const int tcol = (tid & 1) << 4;   // 0 or 16 (halves)

  f32x8 acc[2][4] = {};

  for (int k0 = 0; k0 < K; k0 += 32) {
    u32x4 av0 = *(const u32x4*)(Ab + (long)trow * lda + k0 + tcol);
    u32x4 av1 = *(const u32x4*)(Ab + (long)trow * lda + k0 + tcol + 8);
    u32x4 bv0 = *(const u32x4*)(Bb + (long)trow * ldb + k0 + tcol);
    u32x4 bv1 = *(const u32x4*)(Bb + (long)trow * ldb + k0 + tcol + 8);
    if (k0 + 32 < K) {
      __builtin_prefetch(Ab + (long)trow * lda + k0 + 32 + tcol, 0, 1);
      __builtin_prefetch(Bb + (long)trow * ldb + k0 + 32 + tcol, 0, 1);
    }
    __syncthreads();                               // protect previous frag reads
    *(u32x4*)(sA + trow * 40 + tcol)     = av0;
    *(u32x4*)(sA + trow * 40 + tcol + 8) = av1;
    *(u32x4*)(sB + trow * 40 + tcol)     = bv0;
    *(u32x4*)(sB + trow * 40 + tcol + 8) = bv1;
    __syncthreads();

    // Fragment gathers follow the CDNA5 16-bit WMMA VGPR layouts:
    //  A (16x32): lane m=l15, halves 0..7 = K(8*hi .. 8*hi+7),
    //             halves 8..15 = K(16+8*hi ..)  -> two ds_load_b128
    //  B (32x16): lane n=l15, same K order as A -> two ds_load_b128
    FragU a[2], b[4];
#pragma unroll
    for (int i = 0; i < 2; ++i) {
      const int m = wm * 32 + i * 16 + l15;
      a[i].q[0] = *(const u32x4*)(sA + m * 40 + 8 * hi);
      a[i].q[1] = *(const u32x4*)(sA + m * 40 + 16 + 8 * hi);
    }
#pragma unroll
    for (int j = 0; j < 4; ++j) {
      const int n = wn * 64 + j * 16 + l15;
      b[j].q[0] = *(const u32x4*)(sB + n * 40 + 8 * hi);
      b[j].q[1] = *(const u32x4*)(sB + n * 40 + 16 + 8 * hi);
    }
#pragma unroll
    for (int i = 0; i < 2; ++i)
#pragma unroll
      for (int j = 0; j < 4; ++j)
        acc[i][j] = __builtin_amdgcn_wmma_f32_16x16x32_bf16(
            false, a[i].v, false, b[j].v, (short)0, acc[i][j], false, false);
  }

  const long cOff = (long)(z / cInnerN) * cOuter + (long)(z % cInnerN) * cInner;
#pragma unroll
  for (int i = 0; i < 2; ++i) {
#pragma unroll
    for (int j = 0; j < 4; ++j) {
      const int colL = (int)blockIdx.x * 128 + wn * 64 + j * 16 + l15;
      const float bval = (flags & FLAG_BIAS) ? bias[colL] : 0.0f;
#pragma unroll
      for (int r = 0; r < 8; ++r) {
        // C/D layout: lane l15 = N, VGPR r -> M = r + 8*hi
        const int rowL = (int)blockIdx.y * 128 + wm * 32 + i * 16 + r + 8 * hi;
        float v = acc[i][j][r] + bval;
        if (flags & FLAG_GELU) v = gelu_exact(v);
        const long off = cOff + (long)rowL * ldc + colL;
        if (flags & FLAG_STF32)       Cf[off]  = v;
        else if (flags & FLAG_ADDF32) Cf[off] += v;
        else                          Cb[off]  = f2bf(v);
      }
    }
  }
}

// ---------------------------------------------------------------------------
// LayerNorm over D=1024 (f32 in -> bf16 out), one block per row.
// ---------------------------------------------------------------------------
__global__ __launch_bounds__(256) void ln_bf16_kernel(
    const float* __restrict__ X, const float* __restrict__ w,
    const float* __restrict__ b, unsigned short* __restrict__ Y)
{
  __shared__ float s1[256], s2[256];
  const int row = blockIdx.x, tid = threadIdx.x;
  const float* xr = X + (long)row * 1024;
  float x[4], sum = 0.f, sq = 0.f;
#pragma unroll
  for (int t = 0; t < 4; ++t) { x[t] = xr[tid + 256 * t]; sum += x[t]; sq += x[t] * x[t]; }
  s1[tid] = sum; s2[tid] = sq; __syncthreads();
  for (int o = 128; o > 0; o >>= 1) {
    if (tid < o) { s1[tid] += s1[tid + o]; s2[tid] += s2[tid + o]; }
    __syncthreads();
  }
  const float mean = s1[0] * (1.0f / 1024.0f);
  const float var  = s2[0] * (1.0f / 1024.0f) - mean * mean;
  const float inv  = rsqrtf(var + 1e-5f);
#pragma unroll
  for (int t = 0; t < 4; ++t) {
    const int c = tid + 256 * t;
    Y[(long)row * 1024 + c] = f2bf((x[t] - mean) * inv * w[c] + b[c]);
  }
}

// ---------------------------------------------------------------------------
// RoPE + scale + head relayout:  qraw/kraw (b,s,h*d) -> qh/kh (b,h,s,d) bf16,
// vraw -> vt (b,h,d,s) bf16 (so PV GEMM is K-contiguous over s).
// ---------------------------------------------------------------------------
__global__ __launch_bounds__(256) void qkv_post_kernel(
    const unsigned short* __restrict__ qraw, const unsigned short* __restrict__ kraw,
    const unsigned short* __restrict__ vraw,
    unsigned short* __restrict__ qh, unsigned short* __restrict__ kh,
    unsigned short* __restrict__ vt)
{
  const int row = blockIdx.x;        // b*1024 + s
  const int b   = row >> 10;
  const int s   = row & 1023;
  const int tid = threadIdx.x;
  const float scale = 0.29730177875068026f;   // 128^-0.25
  for (int p = tid; p < 512; p += 256) {      // (h, pair i) pairs
    const int h = p >> 6, i = p & 63;
    const float inv = powf(10000.0f, -(float)i * (1.0f / 64.0f));
    const float ang = (float)s * inv;
    const float c = cosf(ang), sn = sinf(ang);
    const long src = (long)row * 1024 + h * 128 + 2 * i;
    const long dst = ((long)(b * 8 + h) * 1024 + s) * 128 + 2 * i;
    const float q0 = bf2f(qraw[src]), q1 = bf2f(qraw[src + 1]);
    qh[dst]     = f2bf((q0 * c  - q1 * sn) * scale);
    qh[dst + 1] = f2bf((q0 * sn + q1 * c)  * scale);
    const float k0 = bf2f(kraw[src]), k1 = bf2f(kraw[src + 1]);
    kh[dst]     = f2bf((k0 * c  - k1 * sn) * scale);
    kh[dst + 1] = f2bf((k0 * sn + k1 * c)  * scale);
  }
  for (int d = tid; d < 1024; d += 256) {
    const int h = d >> 7, dd = d & 127;
    vt[((long)(b * 8 + h) * 128 + dd) * 1024 + s] = vraw[(long)row * 1024 + d];
  }
}

// ---------------------------------------------------------------------------
// Masked softmax over a score chunk: rows = 64 batches * 128 queries, keys=1024.
// Mask: position t valid iff 4t+3 < x_len[b]  (double stride-2 slicing).
// ---------------------------------------------------------------------------
__global__ __launch_bounds__(256) void softmax_kernel(
    const float* __restrict__ S, unsigned short* __restrict__ P,
    const int* __restrict__ xlen)
{
  __shared__ float red[256];
  const int R = blockIdx.x, tid = threadIdx.x;
  const int b = R >> 10;                      // R = (b*8+h)*128 + q
  const int L = xlen[b];
  const float* sr = S + (long)R * 1024;
  float v[4], mx = -3.4e38f;
#pragma unroll
  for (int t = 0; t < 4; ++t) {
    const int k = tid + 256 * t;
    float x = sr[k];
    if (4 * k + 3 >= L) x += -1e10f;
    v[t] = x; mx = fmaxf(mx, x);
  }
  red[tid] = mx; __syncthreads();
  for (int o = 128; o > 0; o >>= 1) { if (tid < o) red[tid] = fmaxf(red[tid], red[tid + o]); __syncthreads(); }
  mx = red[0]; __syncthreads();
  float sum = 0.f;
#pragma unroll
  for (int t = 0; t < 4; ++t) { v[t] = expf(v[t] - mx); sum += v[t]; }
  red[tid] = sum; __syncthreads();
  for (int o = 128; o > 0; o >>= 1) { if (tid < o) red[tid] += red[tid + o]; __syncthreads(); }
  const float inv = 1.0f / red[0];
#pragma unroll
  for (int t = 0; t < 4; ++t) P[(long)R * 1024 + tid + 256 * t] = f2bf(v[t] * inv);
}

// im2col for conv1: Xcol[(b,t), c*3+r] = x[b, c, 2t-1+r] (zero-padded), bf16.
__global__ __launch_bounds__(128) void im2col1_kernel(
    const float* __restrict__ x, unsigned short* __restrict__ Xcol)
{
  const int row = blockIdx.x;                 // b*2048 + t
  const int b = row >> 11, t = row & 2047;
  const int c = threadIdx.x;
  const float* xb = x + ((long)b * 128 + c) * 4096;
#pragma unroll
  for (int r = 0; r < 3; ++r) {
    const int pos = 2 * t - 1 + r;
    const float v = (pos >= 0 && pos < 4096) ? xb[pos] : 0.0f;
    Xcol[(long)row * 384 + c * 3 + r] = f2bf(v);
  }
}

__global__ void cast_f32_bf16_kernel(const float* __restrict__ s,
                                     unsigned short* __restrict__ d, long n) {
  long i = (long)blockIdx.x * blockDim.x + threadIdx.x;
  const long stride = (long)gridDim.x * blockDim.x;
  for (; i < n; i += stride) d[i] = f2bf(s[i]);
}

// conv2 weight permute: dst[d, r*1024+c] = w[(d*1024+c)*3 + r]  (K = pos-major).
__global__ void cast_conv2_kernel(const float* __restrict__ w,
                                  unsigned short* __restrict__ d) {
  long i = (long)blockIdx.x * blockDim.x + threadIdx.x;
  const long stride = (long)gridDim.x * blockDim.x;
  for (; i < 3145728L; i += stride) {
    const long dd = i / 3072, rem = i % 3072;
    const long r = rem / 1024, c = rem % 1024;
    d[i] = f2bf(w[(dd * 1024 + c) * 3 + r]);
  }
}

__global__ void ylen_kernel(const int* __restrict__ xlen, int* __restrict__ out) {
  const int b = threadIdx.x;
  if (b < 8) { int y = (xlen[b] + 1) / 2; out[b] = (y + 1) / 2; }
}

// ---------------------------------------------------------------------------
extern "C" void kernel_launch(void* const* d_in, const int* in_sizes, int n_in,
                              void* d_out, int out_size, void* d_ws, size_t ws_size,
                              hipStream_t stream)
{
  (void)in_sizes; (void)n_in; (void)out_size; (void)ws_size;

  const float* x         = (const float*)d_in[0];
  const int*   x_len     = (const int*)  d_in[1];
  const float* conv1_b   = (const float*)d_in[3];
  const float* conv2_b   = (const float*)d_in[5];
  const float* attn_ln_w = (const float*)d_in[6];
  const float* attn_ln_b = (const float*)d_in[7];
  const float* q_b       = (const float*)d_in[9];
  const float* v_b       = (const float*)d_in[12];
  const float* out_b     = (const float*)d_in[14];
  const float* mlp_ln_w  = (const float*)d_in[15];
  const float* mlp_ln_b  = (const float*)d_in[16];
  const float* mlp1_b    = (const float*)d_in[18];
  const float* mlp2_b    = (const float*)d_in[20];

  float* carry = (float*)d_out;                         // (B,S,D) f32 output/residual
  int* ylen_out = (int*)d_out + 8L * 1024 * 1024;

  // ---- workspace layout (bump allocator, 256B aligned) --------------------
  char* wsB = (char*)d_ws;
  size_t cur = 0;
  auto take16 = [&](size_t elems) -> unsigned short* {
    unsigned short* p = (unsigned short*)(wsB + cur);
    cur = (cur + elems * 2 + 255) & ~(size_t)255; return p;
  };
  auto take32 = [&](size_t elems) -> float* {
    float* p = (float*)(wsB + cur);
    cur = (cur + elems * 4 + 255) & ~(size_t)255; return p;
  };
  unsigned short* wb_c1 = take16(393216);        // conv1 (1024x384)
  unsigned short* wb_c2 = take16(3145728);       // conv2 permuted (1024x3072)
  unsigned short* wb_q  = take16(6L * 1048576);
  unsigned short* wb_k  = take16(6L * 1048576);
  unsigned short* wb_v  = take16(6L * 1048576);
  unsigned short* wb_o  = take16(6L * 1048576);
  unsigned short* wb_m1 = take16(6L * 4194304);
  unsigned short* wb_m2 = take16(6L * 4194304);
  unsigned short* actA  = take16(8388608);       // LN out / attention out (8192x1024)
  unsigned short* qraw  = take16(8388608);
  unsigned short* kraw  = take16(8388608);
  unsigned short* vraw  = take16(8388608);
  unsigned short* qh    = take16(8388608);       // (b,h,s,d)
  unsigned short* kh    = take16(8388608);       // (b,h,s,d)
  unsigned short* vt    = take16(8388608);       // (b,h,d,s)
  unsigned short* probsB= take16(8388608);       // chunk probs bf16
  unsigned short* hidB  = take16(33554432);      // MLP hidden (8192x4096) bf16
  float*          scoresF = take32(8388608);     // chunk scores f32 (64*128*1024)
  // conv scratch overlaid on hidB (only used before the transformer layers)
  unsigned short* Xcol = hidB;                   // 16384 x 384
  unsigned short* h1p  = hidB + 6291456;         // 8 x 2050 x 1024 (zero-padded)

  auto gemm = [&](const unsigned short* A, long aStride, int lda,
                  const unsigned short* Bw, long bStride, int ldb,
                  float* Cf, unsigned short* Cb,
                  long cOuter, long cInner, int cInnerN, int ldc,
                  int M, int N, int K, int batches,
                  const float* bias, int flags) {
    dim3 grid(N / 128, M / 128, batches);
    gemm_bf16_kernel<<<grid, dim3(256), 0, stream>>>(
        A, aStride, lda, Bw, bStride, ldb, Cf, Cb,
        cOuter, cInner, cInnerN, ldc, K, bias, flags);
  };
  auto cast = [&](const float* src, unsigned short* dst, long n) {
    cast_f32_bf16_kernel<<<2048, 256, 0, stream>>>(src, dst, n);
  };

  // ---- weight casts (deterministic; redone every call) --------------------
  cast((const float*)d_in[2],  wb_c1, 393216);
  cast_conv2_kernel<<<2048, 256, 0, stream>>>((const float*)d_in[4], wb_c2);
  cast((const float*)d_in[8],  wb_q, 6L * 1048576);
  cast((const float*)d_in[10], wb_k, 6L * 1048576);
  cast((const float*)d_in[11], wb_v, 6L * 1048576);
  cast((const float*)d_in[13], wb_o, 6L * 1048576);
  cast((const float*)d_in[17], wb_m1, 6L * 4194304);
  cast((const float*)d_in[19], wb_m2, 6L * 4194304);

  // ---- conv stem ----------------------------------------------------------
  hipMemsetAsync(h1p, 0, (size_t)16793600 * 2, stream);   // zero pad rows
  im2col1_kernel<<<16384, 128, 0, stream>>>(x, Xcol);
  // conv1: per-batch M=2048, K=384 -> h1p interior rows (offset +1 row)
  gemm(Xcol, 786432, 384, wb_c1, 0, 384, nullptr, h1p + 1024,
       2099200, 0, 1, 1024, 2048, 1024, 384, 8, conv1_b, FLAG_BIAS | FLAG_GELU);
  // conv2: overlapping rows (lda=2048, K=3072) -> carry f32
  gemm(h1p, 2099200, 2048, wb_c2, 0, 3072, carry, nullptr,
       1048576, 0, 1, 1024, 1024, 1024, 3072, 8, conv2_b,
       FLAG_BIAS | FLAG_GELU | FLAG_STF32);

  // ---- transformer layers -------------------------------------------------
  for (int l = 0; l < 6; ++l) {
    const unsigned short* Wq  = wb_q  + (long)l * 1048576;
    const unsigned short* Wk  = wb_k  + (long)l * 1048576;
    const unsigned short* Wv  = wb_v  + (long)l * 1048576;
    const unsigned short* Wo  = wb_o  + (long)l * 1048576;
    const unsigned short* Wm1 = wb_m1 + (long)l * 4194304;
    const unsigned short* Wm2 = wb_m2 + (long)l * 4194304;

    ln_bf16_kernel<<<8192, 256, 0, stream>>>(carry, attn_ln_w + l * 1024,
                                             attn_ln_b + l * 1024, actA);
    gemm(actA, 0, 1024, Wq, 0, 1024, nullptr, qraw, 0, 0, 1, 1024,
         8192, 1024, 1024, 1, q_b + l * 1024, FLAG_BIAS);
    gemm(actA, 0, 1024, Wk, 0, 1024, nullptr, kraw, 0, 0, 1, 1024,
         8192, 1024, 1024, 1, nullptr, 0);
    gemm(actA, 0, 1024, Wv, 0, 1024, nullptr, vraw, 0, 0, 1, 1024,
         8192, 1024, 1024, 1, v_b + l * 1024, FLAG_BIAS);
    qkv_post_kernel<<<8192, 256, 0, stream>>>(qraw, kraw, vraw, qh, kh, vt);

    for (int qc = 0; qc < 8; ++qc) {          // 128-query chunks, 64 (b,h) batches
      gemm(qh + (long)qc * 16384, 131072, 128, kh, 131072, 128,
           scoresF, nullptr, 131072, 0, 1, 1024,
           128, 1024, 128, 64, nullptr, FLAG_STF32);
      softmax_kernel<<<8192, 256, 0, stream>>>(scoresF, probsB, x_len);
      // PV: C scatter -> actA row (b*1024 + qc*128 + q), col h*128 + d
      gemm(probsB, 131072, 1024, vt, 131072, 1024,
           nullptr, actA + (long)qc * 131072, 1048576, 128, 8, 1024,
           128, 128, 1024, 64, nullptr, 0);
    }

    gemm(actA, 0, 1024, Wo, 0, 1024, carry, nullptr, 0, 0, 1, 1024,
         8192, 1024, 1024, 1, out_b + l * 1024, FLAG_BIAS | FLAG_ADDF32);

    ln_bf16_kernel<<<8192, 256, 0, stream>>>(carry, mlp_ln_w + l * 1024,
                                             mlp_ln_b + l * 1024, actA);
    gemm(actA, 0, 1024, Wm1, 0, 1024, nullptr, hidB, 0, 0, 1, 4096,
         8192, 4096, 1024, 1, mlp1_b + l * 4096, FLAG_BIAS | FLAG_GELU);
    gemm(hidB, 0, 4096, Wm2, 0, 4096, carry, nullptr, 0, 0, 1, 1024,
         8192, 1024, 4096, 1, mlp2_b + l * 1024, FLAG_BIAS | FLAG_ADDF32);
  }

  ylen_kernel<<<1, 32, 0, stream>>>(x_len, ylen_out);
}
